// Attention_Layer_21131239096479
// MI455X (gfx1250) — matrix operand order repaired and checked by reading, not compiled, not run on hardware
//
#include <hip/hip_runtime.h>

typedef __attribute__((ext_vector_type(16))) __bf16 v16bf;
typedef __attribute__((ext_vector_type(8)))  __bf16 v8bf;
typedef __attribute__((ext_vector_type(8)))  float  v8f;

#define NNODES 10000
#define NEDGES 320000
#define CZ 256
#define CE 64
#define CF 576      // 2*CZ + CE
#define NHEADS 8
#define TILE_E 32   // edges per block in the fused edge kernel (2 M-tiles)

// ---------------------------------------------------------------------------
// Pack fp32 weight matrix W[K, NC] (row-major) into bf16 WMMA B-fragment order.
// B 32x16 fragment layout (wave32): lane = N column (lane%16), lanes 0-15 hold
// K=0..7,16..23, lanes 16-31 hold K=8..15,24..31, 16 contiguous bf16 per lane.
// ---------------------------------------------------------------------------
__global__ void pack_b_kernel(const float* __restrict__ W, __bf16* __restrict__ out,
                              int K, int NC) {
  int idx = blockIdx.x * 256 + threadIdx.x;
  int total = K * NC;
  if (idx >= total) return;
  int t    = idx & 15;
  int lane = (idx >> 4) & 31;
  int tile = idx >> 9;            // kt * NT + nt
  int NT = NC >> 4;
  int nt = tile % NT;
  int kt = tile / NT;
  int k = kt * 32 + (lane >> 4) * 8 + (t & 7) + (t >> 3) * 16;
  int n = nt * 16 + (lane & 15);
  out[idx] = (__bf16)W[(size_t)k * NC + n];
}

__global__ void init_kernel(float* __restrict__ outAgg, float* __restrict__ denom,
                            unsigned* __restrict__ segmax) {
  size_t i = (size_t)blockIdx.x * 256 + threadIdx.x;
  if (i < (size_t)NNODES * CZ) outAgg[i] = 0.f;
  if (i < (size_t)NNODES * NHEADS) { denom[i] = 0.f; segmax[i] = 0u; }
}

// ---------------------------------------------------------------------------
// Generic fused GEMM: C[M,NC] = act(A[M,K] @ Bpacked + bias)
// act: 0 = none, 1 = silu.  M%16==0, K%32==0, NC%16==0, K<=512.
// Block = 256 threads = 8 waves; 16-row M-tile staged in LDS as bf16.
// ---------------------------------------------------------------------------
__global__ __launch_bounds__(256) void gemm_kernel(
    const float* __restrict__ A, const __bf16* __restrict__ Bp,
    const float* __restrict__ bias, float* __restrict__ C,
    int K, int NC, int act) {
  __shared__ __bf16 sA[16 * 512];
  const int tid = threadIdx.x;
  const int m0 = blockIdx.x * 16;
  for (int i = tid; i < 16 * K; i += 256) {
    int r = i / K, c = i - r * K;
    sA[i] = (__bf16)A[(size_t)(m0 + r) * K + c];
  }
  __syncthreads();
  const int lane = tid & 31, w = tid >> 5;
  const int l16 = lane >> 4, lm = lane & 15;
  const int NT = NC >> 4, KT = K >> 5;
  const __bf16* arow = &sA[lm * K];
  for (int nt = w; nt < NT; nt += 8) {
    v8f acc = {};
    for (int kt = 0; kt < KT; ++kt) {
      union { v16bf v; v8bf h2[2]; } a;
      a.h2[0] = *(const v8bf*)(arow + kt * 32 + l16 * 8);
      a.h2[1] = *(const v8bf*)(arow + kt * 32 + 16 + l16 * 8);
      v16bf b = *(const v16bf*)(Bp + (((size_t)kt * NT + nt) * 32 + lane) * 16);
      acc = __builtin_amdgcn_wmma_f32_16x16x32_bf16(false, a.v, false, b,
                                                    (short)0, acc, false, false);
    }
    const int col = nt * 16 + lm;
    const float bc = bias[col];
#pragma unroll
    for (int r = 0; r < 8; ++r) {
      const int mrow = r + l16 * 8;
      float v = acc[r] + bc;
      if (act) v = v * (1.f / (1.f + __expf(-v)));
      C[(size_t)(m0 + mrow) * NC + col] = v;
    }
  }
}

// ---------------------------------------------------------------------------
// Fused per-edge kernel, 32 edges (two 16-row M-tiles) per block:
// gather [edge_attr | x[src] | x[dst]] -> layernorm -> bf16 LDS ->
// K GEMM (waves 0-3, folded straight into per-head logits) and V GEMM
// (waves 4-7, stored bf16). Each B fragment is reused for both M-tiles,
// doubling arithmetic intensity vs. L2. Then scaled logits + atomic seg-max.
// ---------------------------------------------------------------------------
__global__ __launch_bounds__(256) void edge_kernel(
    const float* __restrict__ x, const int* __restrict__ edge_index,
    const float* __restrict__ edge_attr,
    const float* __restrict__ ln_gamma, const float* __restrict__ ln_beta,
    const float* __restrict__ query,
    const __bf16* __restrict__ WkP, const __bf16* __restrict__ WvP,
    const float* __restrict__ bk, const float* __restrict__ bv,
    __bf16* __restrict__ val, float* __restrict__ logits,
    unsigned* __restrict__ segmax) {
  __shared__ __bf16 sX[TILE_E * CF];        // layernormed features, bf16 (36 KB)
  __shared__ float  sQ[TILE_E * CZ];        // query rows of dst nodes   (32 KB)
  __shared__ float  sLog[TILE_E * NHEADS];  // per-edge per-head logits
  __shared__ int    sDst[TILE_E];

  const int tid = threadIdx.x;
  const int lane = tid & 31, w = tid >> 5;
  const int e0 = blockIdx.x * TILE_E;

  sLog[tid] = 0.f;                          // TILE_E*NHEADS == 256 == blockDim

  // ---- phase 1: gather + layernorm (each wave handles 4 rows) --------------
  for (int rr = 0; rr < 4; ++rr) {
    const int row = w * 4 + rr;
    const int e = e0 + row;
    const int src = edge_index[e];
    const int dst = edge_index[NEDGES + e];
    if (lane == 0) sDst[row] = dst;
    float vals[18];
    float s = 0.f, s2 = 0.f;
    int j = 0;
    for (int i = lane; i < CF; i += 32, ++j) {
      float v;
      if (i < CE)            v = edge_attr[(size_t)e * CE + i];
      else if (i < CE + CZ)  v = x[(size_t)src * CZ + (i - CE)];
      else                   v = x[(size_t)dst * CZ + (i - CE - CZ)];
      vals[j] = v;
      s += v; s2 += v * v;
    }
#pragma unroll
    for (int m = 16; m; m >>= 1) {
      s  += __shfl_xor(s,  m, 32);
      s2 += __shfl_xor(s2, m, 32);
    }
    const float mu = s * (1.f / CF);
    const float rstd = rsqrtf(s2 * (1.f / CF) - mu * mu + 1e-5f);
    j = 0;
    for (int i = lane; i < CF; i += 32, ++j)
      sX[row * CF + i] = (__bf16)((vals[j] - mu) * rstd * ln_gamma[i] + ln_beta[i]);
    for (int i = lane; i < CZ; i += 32)
      sQ[row * CZ + i] = query[(size_t)dst * CZ + i];
  }
  __syncthreads();

  // ---- phase 2: K GEMM (waves 0-3) / V GEMM (waves 4-7), 2 M-tiles ---------
  const bool isKey = (w < 4);
  const __bf16* Bp = isKey ? WkP : WvP;
  const float* bias = isKey ? bk : bv;
  const int l16 = lane >> 4, lm = lane & 15;
  const __bf16* arow0 = &sX[lm * CF];
  const __bf16* arow1 = &sX[(16 + lm) * CF];
  for (int jnt = 0; jnt < 4; ++jnt) {
    const int nt = (w & 3) * 4 + jnt;   // key wave w owns heads 2w,2w+1 exclusively
    v8f acc0 = {}, acc1 = {};
    for (int kt = 0; kt < CF / 32; ++kt) {
      v16bf b = *(const v16bf*)(Bp + (((size_t)kt * (CZ / 16) + nt) * 32 + lane) * 16);
      union { v16bf v; v8bf h2[2]; } a0, a1;
      a0.h2[0] = *(const v8bf*)(arow0 + kt * 32 + l16 * 8);
      a0.h2[1] = *(const v8bf*)(arow0 + kt * 32 + 16 + l16 * 8);
      a1.h2[0] = *(const v8bf*)(arow1 + kt * 32 + l16 * 8);
      a1.h2[1] = *(const v8bf*)(arow1 + kt * 32 + 16 + l16 * 8);
      acc0 = __builtin_amdgcn_wmma_f32_16x16x32_bf16(false, a0.v, false, b,
                                                     (short)0, acc0, false, false);
      acc1 = __builtin_amdgcn_wmma_f32_16x16x32_bf16(false, a1.v, false, b,
                                                     (short)0, acc1, false, false);
    }
    const int col = nt * 16 + lm;
    const float bc = bias[col];
    if (isKey) {
      const int h = nt >> 1;
#pragma unroll
      for (int mt = 0; mt < 2; ++mt) {
#pragma unroll
        for (int r = 0; r < 8; ++r) {
          const int mrow = mt * 16 + r + l16 * 8;    // edge row of this D element
          const float kv = (mt ? acc1[r] : acc0[r]) + bc;
          float p = kv * sQ[mrow * CZ + col];
#pragma unroll
          for (int m = 8; m; m >>= 1) p += __shfl_xor(p, m, 32);  // reduce 16 cols
          if (lm == 0) sLog[mrow * NHEADS + h] += p; // lanes 0/16 -> rows r, r+8
        }
      }
    } else {
#pragma unroll
      for (int mt = 0; mt < 2; ++mt) {
#pragma unroll
        for (int r = 0; r < 8; ++r) {
          const int mrow = mt * 16 + r + l16 * 8;
          const float vv = (mt ? acc1[r] : acc0[r]) + bc;
          val[(size_t)(e0 + mrow) * CZ + col] = (__bf16)vv;
        }
      }
    }
  }
  __syncthreads();

  // ---- phase 3: write logits, atomic segment max (256 = 32 edges x 8 heads)
  {
    const int le = tid >> 3, h = tid & 7;
    const float lg = sLog[tid] * 0.17677669529663689f;  // 1/sqrt(32)
    logits[(size_t)(e0 + le) * NHEADS + h] = lg;
    unsigned bits = __float_as_uint(lg);
    unsigned enc = (bits & 0x80000000u) ? ~bits : (bits | 0x80000000u);
    atomicMax(&segmax[(size_t)sDst[le] * NHEADS + h], enc);
  }
}

// exp(logit - segmax[dst]) and atomic denom accumulation
__global__ void softmax_edge_kernel(const float* __restrict__ logits,
                                    const unsigned* __restrict__ segmax,
                                    const int* __restrict__ dstArr,
                                    float* __restrict__ ex,
                                    float* __restrict__ denom) {
  size_t idx = (size_t)blockIdx.x * 256 + threadIdx.x;
  if (idx >= (size_t)NEDGES * NHEADS) return;
  int h = idx & 7;
  size_t e = idx >> 3;
  int dst = dstArr[e];
  unsigned u = segmax[(size_t)dst * NHEADS + h];
  unsigned bits = (u & 0x80000000u) ? (u & 0x7FFFFFFFu) : ~u;
  float m = __uint_as_float(bits);
  if (!__builtin_isfinite(m)) m = 0.f;     // matches reference isfinite clamp
  float v = __expf(logits[idx] - m);
  ex[idx] = v;
  atomicAdd(&denom[(size_t)dst * NHEADS + h], v);
}

// out[dst, c] += (ex/denom) * value[e, c]
__global__ void aggregate_kernel(const __bf16* __restrict__ val,
                                 const float* __restrict__ ex,
                                 const float* __restrict__ denom,
                                 const int* __restrict__ dstArr,
                                 float* __restrict__ outAgg) {
  size_t idx = (size_t)blockIdx.x * 256 + threadIdx.x;
  if (idx >= (size_t)NEDGES * CZ) return;
  int c = idx & (CZ - 1);
  size_t e = idx >> 8;
  int h = c >> 5;
  int dst = dstArr[e];
  float alpha = ex[e * NHEADS + h] / denom[(size_t)dst * NHEADS + h];
  atomicAdd(&outAgg[(size_t)dst * CZ + c], alpha * (float)val[idx]);
}

extern "C" void kernel_launch(void* const* d_in, const int* in_sizes, int n_in,
                              void* d_out, int out_size, void* d_ws, size_t ws_size,
                              hipStream_t stream) {
  const float* x         = (const float*)d_in[0];
  const int*   edge_idx  = (const int*)d_in[1];
  const float* edge_attr = (const float*)d_in[2];
  const float* ln_gamma  = (const float*)d_in[3];
  const float* ln_beta   = (const float*)d_in[4];
  const float* Wq = (const float*)d_in[5];
  const float* bq = (const float*)d_in[6];
  const float* Wk = (const float*)d_in[7];
  const float* bk = (const float*)d_in[8];
  const float* Wv = (const float*)d_in[9];
  const float* bv = (const float*)d_in[10];
  const float* W1 = (const float*)d_in[11];
  const float* b1 = (const float*)d_in[12];
  const float* W2 = (const float*)d_in[13];
  const float* b2 = (const float*)d_in[14];
  float* y = (float*)d_out;

  char* ws = (char*)d_ws;
  size_t off = 0;
  auto alloc = [&](size_t bytes) -> char* {
    char* p = ws + off;
    off += (bytes + 255) & ~(size_t)255;
    return p;
  };

  __bf16*   WqP    = (__bf16*)alloc((size_t)CZ * CZ * 2);
  __bf16*   WkP    = (__bf16*)alloc((size_t)CF * CZ * 2);
  __bf16*   WvP    = (__bf16*)alloc((size_t)CF * CZ * 2);
  __bf16*   W1P    = (__bf16*)alloc((size_t)CZ * 2 * CZ * 2);
  __bf16*   W2P    = (__bf16*)alloc((size_t)2 * CZ * CZ * 2);
  float*    query  = (float*)alloc((size_t)NNODES * CZ * 4);
  unsigned* segmax = (unsigned*)alloc((size_t)NNODES * NHEADS * 4);
  float*    denom  = (float*)alloc((size_t)NNODES * NHEADS * 4);
  float*    logits = (float*)alloc((size_t)NEDGES * NHEADS * 4);
  float*    exv    = (float*)alloc((size_t)NEDGES * NHEADS * 4);
  __bf16*   val    = (__bf16*)alloc((size_t)NEDGES * CZ * 2);
  float*    outAgg = (float*)alloc((size_t)NNODES * CZ * 4);
  float*    hbuf   = (float*)alloc((size_t)NNODES * 2 * CZ * 4);
  (void)ws_size; (void)in_sizes; (void)n_in; (void)out_size;

  // 1. pack all weights to bf16 WMMA B-fragment order
  pack_b_kernel<<<(CZ * CZ + 255) / 256, 256, 0, stream>>>(Wq, WqP, CZ, CZ);
  pack_b_kernel<<<(CF * CZ + 255) / 256, 256, 0, stream>>>(Wk, WkP, CF, CZ);
  pack_b_kernel<<<(CF * CZ + 255) / 256, 256, 0, stream>>>(Wv, WvP, CF, CZ);
  pack_b_kernel<<<(CZ * 2 * CZ + 255) / 256, 256, 0, stream>>>(W1, W1P, CZ, 2 * CZ);
  pack_b_kernel<<<(2 * CZ * CZ + 255) / 256, 256, 0, stream>>>(W2, W2P, 2 * CZ, CZ);

  // 2. zero accumulators / init segment max
  init_kernel<<<(NNODES * CZ + 255) / 256, 256, 0, stream>>>(outAgg, denom, segmax);

  // 3. query = x @ Wq + bq
  gemm_kernel<<<NNODES / 16, 256, 0, stream>>>(x, WqP, bq, query, CZ, CZ, 0);

  // 4. fused edge pipeline: gather+LN+K/V GEMM+logits+segmax
  edge_kernel<<<NEDGES / TILE_E, 256, 0, stream>>>(x, edge_idx, edge_attr, ln_gamma,
                                                   ln_beta, query, WkP, WvP, bk, bv,
                                                   val, logits, segmax);

  // 5. segment softmax numerator + denominator
  softmax_edge_kernel<<<(NEDGES * NHEADS + 255) / 256, 256, 0, stream>>>(
      logits, segmax, edge_idx + NEDGES, exv, denom);

  // 6. weighted segment sum of values
  aggregate_kernel<<<(unsigned)(((size_t)NEDGES * CZ + 255) / 256), 256, 0, stream>>>(
      val, exv, denom, edge_idx + NEDGES, outAgg);

  // 7. FFN: h = silu(out @ W1 + b1); y = h @ W2 + b2
  gemm_kernel<<<NNODES / 16, 256, 0, stream>>>(outAgg, W1P, b1, hbuf, CZ, 2 * CZ, 1);
  gemm_kernel<<<NNODES / 16, 256, 0, stream>>>(hbuf, W2P, b2, y, 2 * CZ, CZ, 0);
}
